// MoELayer_26242250179174
// MI455X (gfx1250) — compile-verified, matrix operand order checked by reading, not run on hardware
//
#include <hip/hip_runtime.h>
#include <hip/hip_bf16.h>

typedef __attribute__((ext_vector_type(16))) _Float16 v16h;
typedef __attribute__((ext_vector_type(8)))  _Float16 v8h;
typedef __attribute__((ext_vector_type(4)))  _Float16 v4h;
typedef __attribute__((ext_vector_type(2)))  _Float16 v2h;
typedef __attribute__((ext_vector_type(8)))  float    v8f;
typedef __attribute__((ext_vector_type(4)))  float    v4f;
typedef __attribute__((ext_vector_type(4)))  unsigned u32x4;
typedef __attribute__((ext_vector_type(8)))  int      i32x8;

#define BB   256   // batch
#define NTOK 64    // tokens per batch (incl. skipped token 0)
#define NA   63    // active tokens (1..63)
#define DD   128   // model dim (K of GEMM1)
#define HH   256   // hidden dim (N of GEMM1, K of GEMM2)
#define EE   8     // experts
#define T2   160   // output cols (N of GEMM2)

// padded row strides (in halfs): +8 halfs = 16B => 4-dword bank skew
#define XS   (DD + 8)   // 136
#define HS   (HH + 8)   // 264
#define W1S  (DD + 8)   // 136  (w1^T rows: n in [0,256), k in [0,128))
#define W2S  (HH + 8)   // 264  (w2^T rows: n in [0,160), k in [0,256))

#define W1T_ESTRIDE (HH * W1S)          // 34816 halfs per expert
#define W2T_ESTRIDE (T2 * W2S)          // 42240 halfs per expert
#define SW_HALFS    42240               // max(W1T_ESTRIDE, W2T_ESTRIDE)
#define LEN1_8B     ((W1T_ESTRIDE * 2) / 8)   // 8704  (fits in 16-bit tile_dim0)
#define LEN2_8B     ((W2T_ESTRIDE * 2) / 8)   // 10560 (fits in 16-bit tile_dim0)

// workspace layout (bytes)
#define WS_TOPK_IDX   0
#define WS_TOPK_GATE  (BB * 2 * sizeof(int))
#define WS_W1T        4096
#define WS_W2T        (WS_W1T + EE * W1T_ESTRIDE * sizeof(_Float16))  // 4096+557056

// ---------------------------------------------------------------------------
// Kernel 0: one-shot weight prep -- f32 -> f16, transposed to N-major rows
// with the bank-padded stride the LDS tiles use, so per-block staging is a
// raw byte copy (TDM-able).
// ---------------------------------------------------------------------------
__global__ __launch_bounds__(256) void moe_prep_weights(
    const float* __restrict__ w1,   // (E, 128, 256)
    const float* __restrict__ w2,   // (E, 256, 160)
    _Float16* __restrict__ w1t,     // (E, 256, W1S)
    _Float16* __restrict__ w2t)     // (E, 160, W2S)
{
    const int idx = blockIdx.x * 256 + threadIdx.x;
    const int total1 = EE * HH * (DD / 2);   // 131072 half-pairs
    const int total2 = EE * T2 * (HH / 2);   // 163840 half-pairs
    if (idx < total1) {
        const int e   = idx / (HH * (DD / 2));
        const int rem = idx % (HH * (DD / 2));
        const int kq  = rem / HH;
        const int n   = rem % HH;
        const int k0  = kq * 2;
        const float* W = w1 + (size_t)e * DD * HH;
        v2h pk;
        pk[0] = (_Float16)W[(size_t)k0 * HH + n];
        pk[1] = (_Float16)W[(size_t)(k0 + 1) * HH + n];
        *(v2h*)(w1t + ((size_t)e * HH + n) * W1S + k0) = pk;
    } else if (idx < total1 + total2) {
        const int id2 = idx - total1;
        const int e   = id2 / (T2 * (HH / 2));
        const int rem = id2 % (T2 * (HH / 2));
        const int kq  = rem / T2;
        const int n   = rem % T2;
        const int k0  = kq * 2;
        const float* W = w2 + (size_t)e * HH * T2;
        v2h pk;
        pk[0] = (_Float16)W[(size_t)k0 * T2 + n];
        pk[1] = (_Float16)W[(size_t)(k0 + 1) * T2 + n];
        *(v2h*)(w2t + ((size_t)e * T2 + n) * W2S + k0) = pk;
    }
}

// ---------------------------------------------------------------------------
// Kernel 1: router -- mean pool, logits, softmax, top-2 gates
// ---------------------------------------------------------------------------
__global__ __launch_bounds__(64) void moe_router_kernel(
    const float* __restrict__ x,
    const float* __restrict__ rw,
    const float* __restrict__ rb,
    int*   __restrict__ topk_idx,
    float* __restrict__ topk_gate)
{
    __shared__ float pooled[DD];
    __shared__ float logits[EE];
    const int b = blockIdx.x;
    const int t = threadIdx.x;

    for (int d = t; d < DD; d += 64) {
        float s = 0.0f;
        for (int r = 1; r < NTOK; ++r)
            s += x[((size_t)b * NTOK + r) * DD + d];
        pooled[d] = s * (1.0f / (float)NA);
    }
    __syncthreads();

    if (t < EE) {
        float s = rb[t];
        for (int d = 0; d < DD; ++d)
            s += pooled[d] * rw[d * EE + t];
        logits[t] = s;
    }
    __syncthreads();

    if (t == 0) {
        float mx = logits[0];
        for (int e = 1; e < EE; ++e) mx = fmaxf(mx, logits[e]);
        float ex[EE], sum = 0.0f;
        for (int e = 0; e < EE; ++e) { ex[e] = __expf(logits[e] - mx); sum += ex[e]; }
        float inv = 1.0f / sum;
        int i0 = 0;
        for (int e = 1; e < EE; ++e) if (ex[e] > ex[i0]) i0 = e;
        int i1 = (i0 == 0) ? 1 : 0;
        for (int e = 0; e < EE; ++e) if (e != i0 && ex[e] > ex[i1]) i1 = e;
        topk_idx[b * 2 + 0]  = i0;
        topk_idx[b * 2 + 1]  = i1;
        topk_gate[b * 2 + 0] = ex[i0] * inv;
        topk_gate[b * 2 + 1] = ex[i1] * inv;
    }
}

// ---------------------------------------------------------------------------
// TDM 1-D copy: global (f16, contiguous) -> LDS, len8 units of 8 bytes.
// D# per CDNA5 ISA 8.3/8.4: group0 = {count=1 | lds_addr | global_addr | type=2},
// group1 = {data_size=8B, tensor_dim0 = tile_dim0 = len8, stride0 = len8}.
// 1-D tensor => D# groups 2/3 are NULL => 2-operand asm form.
// ---------------------------------------------------------------------------
__device__ inline unsigned lds_off_u32(const void* p)
{
    return (unsigned)(unsigned long long)(uintptr_t)p;   // flat LDS addr[31:0]
}

__device__ inline void tdm_load_lds(unsigned lds_off, const _Float16* gsrc,
                                    unsigned len8)
{
    const unsigned long long ga = (unsigned long long)(uintptr_t)gsrc;
    u32x4 g0;
    g0[0] = 1u;                                        // count=1, user mode
    g0[1] = lds_off;                                   // lds_addr
    g0[2] = (unsigned)(ga & 0xFFFFFFFFu);              // global_addr[31:0]
    g0[3] = ((unsigned)(ga >> 32) & 0x01FFFFFFu)       // global_addr[56:32]
            | 0x80000000u;                             // type=2 ("image")
    i32x8 g1;
    g1[0] = (int)(3u << 16);                           // data_size=3 (8B)
    g1[1] = (int)((len8 & 0xFFFFu) << 16);             // tensor_dim0[15:0]
    g1[2] = (int)(len8 >> 16);                         // tensor_dim0[31:16]
    g1[3] = (int)((len8 & 0xFFFFu) << 16);             // tile_dim0
    g1[4] = 0;                                         // tile_dim1/2 unused
    g1[5] = (int)len8;                                 // tensor_dim0_stride
    g1[6] = 0;
    g1[7] = 0;
    asm volatile("tensor_load_to_lds %0, %1" :: "s"(g0), "s"(g1) : "memory");
}

// ---------------------------------------------------------------------------
// WMMA fragment loader from LDS (ISA 7.12.2 16-bit layout, wave32).
// ---------------------------------------------------------------------------
__device__ inline v16h load_frag_lds(const _Float16* base, int stride,
                                     int rbase, int kbase, int lane)
{
    const int r  = rbase + (lane & 15);
    const int kg = (lane >> 4) * 8;
    const _Float16* p = base + r * stride + kbase + kg;
    const v8h lo = *(const v8h*)(p);
    const v8h hi = *(const v8h*)(p + 16);
    v16h a;
#pragma unroll
    for (int i = 0; i < 8; ++i) { a[i] = lo[i]; a[8 + i] = hi[i]; }
    return a;
}

// GEMM1: sH = relu(sX @ sWb + B1); wave covers (mt, nt = (wv&1)*8 .. +7)
__device__ inline void gemm1_tile(const _Float16* sX, const _Float16* sWb,
                                  const float* __restrict__ B1, _Float16* sH,
                                  int wv, int lane, int mt)
{
    v16h afr[4];
#pragma unroll
    for (int ki = 0; ki < 4; ++ki)
        afr[ki] = load_frag_lds(sX, XS, mt * 16, ki * 32, lane);

    const int nt0 = (wv & 1) * 8;
#pragma unroll
    for (int j = 0; j < 8; ++j) {
        const int nt = nt0 + j;
        v8f c = {};
#pragma unroll
        for (int ki = 0; ki < 4; ++ki) {
            v16h bfr = load_frag_lds(sWb, W1S, nt * 16, ki * 32, lane);
            c = __builtin_amdgcn_wmma_f32_16x16x32_f16(
                    false, afr[ki], false, bfr, (short)0, c, false, false);
        }
        const int   ncol  = nt * 16 + (lane & 15);
        const float bias  = B1[ncol];
        const int   rbase = mt * 16 + (lane >> 4) * 8;
#pragma unroll
        for (int r = 0; r < 8; ++r) {
            float v = c[r] + bias;
            v = v > 0.0f ? v : 0.0f;
            sH[(size_t)(rbase + r) * HS + ncol] = (_Float16)v;
        }
    }
}

// GEMM2: acc += g * (sH @ sWb + B2); wave covers (mt, nt = (wv&1)*5 .. +4)
__device__ inline void gemm2_tile(const _Float16* sH, const _Float16* sWb,
                                  const float* __restrict__ B2, float g,
                                  float acc[5][8], int wv, int lane, int mt)
{
    v16h afr[8];
#pragma unroll
    for (int ki = 0; ki < 8; ++ki)
        afr[ki] = load_frag_lds(sH, HS, mt * 16, ki * 32, lane);

    const int nt0 = (wv & 1) * 5;
#pragma unroll
    for (int j = 0; j < 5; ++j) {
        const int nt = nt0 + j;
        v8f c = {};
#pragma unroll
        for (int ki = 0; ki < 8; ++ki) {
            v16h bfr = load_frag_lds(sWb, W2S, nt * 16, ki * 32, lane);
            c = __builtin_amdgcn_wmma_f32_16x16x32_f16(
                    false, afr[ki], false, bfr, (short)0, c, false, false);
        }
        const int   ncol = nt * 16 + (lane & 15);
        const float bias = B2[ncol];
#pragma unroll
        for (int r = 0; r < 8; ++r)
            acc[j][r] += g * (c[r] + bias);
    }
}

// ---------------------------------------------------------------------------
// Kernel 2: per-batch top-2 expert FFN; TDM double-buffered weights.
// Phases: GEMM1(e0,buf0) || DMA w2t[e0]->buf1 ; GEMM2(e0,buf1) || DMA
// w1t[e1]->buf0 ; GEMM1(e1,buf0) || DMA w2t[e1]->buf1 ; GEMM2(e1,buf1).
// ---------------------------------------------------------------------------
__global__ __launch_bounds__(256) void moe_expert_kernel(
    const float*    __restrict__ x,          // (B, 64, 128)
    const _Float16* __restrict__ w1t,        // (E, 256, W1S)
    const float*    __restrict__ b1,         // (E, 256)
    const _Float16* __restrict__ w2t,        // (E, 160, W2S)
    const float*    __restrict__ b2,         // (E, 160)
    const int*      __restrict__ topk_idx,   // (B, 2)
    const float*    __restrict__ topk_gate,  // (B, 2)
    float* __restrict__ out)                 // (B, 63, 160)
{
    __shared__ __align__(16) _Float16 sX[64 * XS];       // ~17 KB
    __shared__ __align__(16) _Float16 sH[64 * HS];       // ~33 KB
    __shared__ __align__(16) _Float16 sW[2][SW_HALFS];   // ~165 KB (ping-pong)

    const int b    = blockIdx.x;
    const int tid  = threadIdx.x;
    const int lane = tid & 31;
    const int wv   = tid >> 5;
    const int mt   = wv >> 1;

    const int   e0 = __builtin_amdgcn_readfirstlane(topk_idx[b * 2 + 0]);
    const int   e1 = __builtin_amdgcn_readfirstlane(topk_idx[b * 2 + 1]);
    const float g0 = topk_gate[b * 2 + 0];
    const float g1 = topk_gate[b * 2 + 1];
    const int   eArr[2] = {e0, e1};
    const float gArr[2] = {g0, g1};

    const unsigned off0 = lds_off_u32(&sW[0][0]);
    const unsigned off1 = lds_off_u32(&sW[1][0]);

    // kick off first weight DMA, overlap with x staging
    if (wv == 0)
        tdm_load_lds(off0, w1t + (size_t)e0 * W1T_ESTRIDE, LEN1_8B);

    // ---- stage xa (tokens 1..63) as f16; pad row 63 with zeros ----
    for (int it = 0; it < (64 * DD) / (256 * 4); ++it) {
        const int idx4 = (tid + it * 256) * 4;
        const int r = idx4 >> 7;
        const int c = idx4 & (DD - 1);
        v4h h4;
        if (r < NA) {
            const v4f f4 = *(const v4f*)(x + ((size_t)b * NTOK + (r + 1)) * DD + c);
#pragma unroll
            for (int q = 0; q < 4; ++q) h4[q] = (_Float16)f4[q];
        } else {
#pragma unroll
            for (int q = 0; q < 4; ++q) h4[q] = (_Float16)0.0f;
        }
        *(v4h*)(sX + (size_t)r * XS + c) = h4;
    }

    if (wv == 0) __builtin_amdgcn_s_wait_tensorcnt(0);
    __syncthreads();

    float acc[5][8];
#pragma unroll
    for (int j = 0; j < 5; ++j)
#pragma unroll
        for (int r = 0; r < 8; ++r) acc[j][r] = 0.0f;

    for (int kk = 0; kk < 2; ++kk) {
        const int   e = eArr[kk];
        const float g = gArr[kk];

        // prefetch this expert's w2^T into buf1 while GEMM1 runs on buf0
        if (wv == 0)
            tdm_load_lds(off1, w2t + (size_t)e * W2T_ESTRIDE, LEN2_8B);

        gemm1_tile(sX, sW[0], b1 + (size_t)e * HH, sH, wv, lane, mt);

        if (wv == 0) __builtin_amdgcn_s_wait_tensorcnt(0);
        __syncthreads();

        // prefetch next expert's w1^T into buf0 while GEMM2 runs on buf1
        if (wv == 0 && kk == 0)
            tdm_load_lds(off0, w1t + (size_t)e1 * W1T_ESTRIDE, LEN1_8B);

        gemm2_tile(sH, sW[1], b2 + (size_t)e * T2, g, acc, wv, lane, mt);

        if (wv == 0) __builtin_amdgcn_s_wait_tensorcnt(0);
        __syncthreads();
    }

    // ---- writeback: rows 0..62 -> out[b, row, :] ----
    const int nt0 = (wv & 1) * 5;
#pragma unroll
    for (int j = 0; j < 5; ++j) {
        const int nt    = nt0 + j;
        const int ncol  = nt * 16 + (lane & 15);
        const int rbase = mt * 16 + (lane >> 4) * 8;
#pragma unroll
        for (int r = 0; r < 8; ++r) {
            const int row = rbase + r;
            if (row < NA)
                out[((size_t)b * NA + row) * T2 + ncol] = acc[j][r];
        }
    }
}

// ---------------------------------------------------------------------------
extern "C" void kernel_launch(void* const* d_in, const int* in_sizes, int n_in,
                              void* d_out, int out_size, void* d_ws, size_t ws_size,
                              hipStream_t stream)
{
    const float* x   = (const float*)d_in[0];
    const float* rw  = (const float*)d_in[1];
    const float* rb  = (const float*)d_in[2];
    const float* w1  = (const float*)d_in[3];
    const float* b1  = (const float*)d_in[4];
    const float* w2  = (const float*)d_in[5];
    const float* b2  = (const float*)d_in[6];

    float* out = (float*)d_out;

    int*      topk_idx  = (int*)((char*)d_ws + WS_TOPK_IDX);
    float*    topk_gate = (float*)((char*)d_ws + WS_TOPK_GATE);
    _Float16* w1t       = (_Float16*)((char*)d_ws + WS_W1T);
    _Float16* w2t       = (_Float16*)((char*)d_ws + WS_W2T);

    const int prep_total  = EE * HH * (DD / 2) + EE * T2 * (HH / 2);
    const int prep_blocks = (prep_total + 255) / 256;   // 1152

    moe_prep_weights<<<prep_blocks, 256, 0, stream>>>(w1, w2, w1t, w2t);
    moe_router_kernel<<<BB, 64, 0, stream>>>(x, rw, rb, topk_idx, topk_gate);
    moe_expert_kernel<<<BB, 256, 0, stream>>>(x, w1t, b1, w2t, b2,
                                              topk_idx, topk_gate, out);
}
